// QCNetBackbone_43954695307305
// MI455X (gfx1250) — matrix operand order repaired and checked by reading, not compiled, not run on hardware
//
#include <hip/hip_runtime.h>
#include <math.h>

// ---------------- constants (from reference) ----------------
#define CB 2
#define CN 48
#define CT 32
#define CM 128
#define CH 128
#define CNH 8
#define CDH 16
#define CF 64
#define CKPL 32
#define CKAA 16
#define CNL 2

#define PI_F 3.14159265358979323846f
#define TWO_PI_F 6.28318530717958647692f

typedef __attribute__((ext_vector_type(16))) _Float16 v16h;
typedef __attribute__((ext_vector_type(8)))  float    v8f;

union FragU { uint4 q[2]; v16h v; };
union H2   { _Float16 h[2]; unsigned u; };

// ================= WMMA GEMM =================
// C(f32[,f16]) = A(f16, MxK row-major) @ B (given as B^T: f16, NxK row-major, K padded)
//   (+bias)(+accum)(+relu).  block = 128 threads (4 waves), tile 64x64, K-chunks of 32.
// Fragments are pairs of contiguous 16B LDS loads (ds_load_b128); staging is
// register-pipelined against the WMMA work.
__global__ __launch_bounds__(128) void k_gemm(
    const _Float16* __restrict__ A, const _Float16* __restrict__ Bt,
    float* __restrict__ Cf, _Float16* __restrict__ Ch,
    const float* __restrict__ bias,
    int Mr, int Nc, int Kd, int lda, int ldb, int accum, int relu)
{
  __shared__ __align__(16) _Float16 sA[64 * 40];
  __shared__ __align__(16) _Float16 sB[64 * 40];

  const int tid  = threadIdx.x;
  const int lane = tid & 31;
  const int wave = tid >> 5;
  const int hi   = lane >> 4;
  const int l15  = lane & 15;
  const int m0   = blockIdx.y * 64;
  const int n0   = blockIdx.x * 64;

  v8f acc[4];
#pragma unroll
  for (int nt = 0; nt < 4; ++nt)
#pragma unroll
    for (int e = 0; e < 8; ++e) acc[nt][e] = 0.0f;

  // staging: 64 rows x 32 halves for both tiles; thread -> (row, 16-half slot)
  const int srow  = tid >> 1;
  const int shalf = (tid & 1) * 16;
  const _Float16* aptr = A  + (size_t)(m0 + srow) * lda + shalf;
  const _Float16* bptr = Bt + (size_t)(n0 + srow) * ldb + shalf;

  uint4 ra0 = *(const uint4*)(aptr);
  uint4 ra1 = *(const uint4*)(aptr + 8);
  uint4 rb0 = *(const uint4*)(bptr);
  uint4 rb1 = *(const uint4*)(bptr + 8);

  for (int k0 = 0; k0 < Kd; k0 += 32) {
    *(uint4*)(sA + srow * 40 + shalf)     = ra0;
    *(uint4*)(sA + srow * 40 + shalf + 8) = ra1;
    *(uint4*)(sB + srow * 40 + shalf)     = rb0;
    *(uint4*)(sB + srow * 40 + shalf + 8) = rb1;
    __syncthreads();

    if (k0 + 32 < Kd) {   // pipeline next chunk into registers
      ra0 = *(const uint4*)(aptr + k0 + 32);
      ra1 = *(const uint4*)(aptr + k0 + 40);
      rb0 = *(const uint4*)(bptr + k0 + 32);
      rb1 = *(const uint4*)(bptr + k0 + 40);
    }

    // A fragment (16x32): halves 0..7 = K(hi*8..+7), 8..15 = K(16+hi*8..+7)
    FragU af;
    {
      const _Float16* ap = sA + (wave * 16 + l15) * 40;
      af.q[0] = *(const uint4*)(ap + hi * 8);
      af.q[1] = *(const uint4*)(ap + 16 + hi * 8);
    }
#pragma unroll
    for (int nt = 0; nt < 4; ++nt) {
      // B fragment (32x16): col = lane&15, halves 0..7 = K(hi*8..+7), 8..15 = K(16+hi*8..+7)
      FragU bf;
      const _Float16* bp = sB + (nt * 16 + l15) * 40;
      bf.q[0] = *(const uint4*)(bp + hi * 8);
      bf.q[1] = *(const uint4*)(bp + 16 + hi * 8);
      acc[nt] = __builtin_amdgcn_wmma_f32_16x16x32_f16(
          false, af.v, false, bf.v, (short)0, acc[nt], false, false);
    }
    __syncthreads();
  }

  // epilogue: C layout -> row = r + 8*hi, col = lane&15 (per 16x16 tile)
#pragma unroll
  for (int nt = 0; nt < 4; ++nt) {
#pragma unroll
    for (int r = 0; r < 8; ++r) {
      int m = m0 + wave * 16 + r + 8 * hi;
      int n = n0 + nt * 16 + l15;
      float v = acc[nt][r];
      if (bias) v += bias[n];
      size_t off = (size_t)m * Nc + n;
      if (accum) v += Cf[off];
      if (relu) v = v > 0.0f ? v : 0.0f;
      if (Cf) Cf[off] = v;
      if (Ch) Ch[off] = (_Float16)v;
    }
  }
}

// ================= small helpers =================
__device__ inline float wrapang(float a) {
  float x = a + PI_F;
  x -= floorf(x * (1.0f / TWO_PI_F)) * TWO_PI_F;
  return x - PI_F;
}
__device__ inline float wred32(float v) {
#pragma unroll
  for (int s = 16; s; s >>= 1) v += __shfl_xor(v, s, 32);
  return v;
}

// f32 (K0 x N) -> f16 (Kp x N), zero pad rows (for A-side operands like x_pl)
__global__ void k_cvt_pad(const float* __restrict__ src, _Float16* __restrict__ dst,
                          int K0, int Kp, int Nc)
{
  int i = blockIdx.x * 256 + threadIdx.x;
  if (i >= Kp * Nc) return;
  int k = i / Nc, n = i - k * Nc;
  dst[i] = (k < K0) ? (_Float16)src[(size_t)k * Nc + n] : (_Float16)0.0f;
}

// f32 weight (K0 x N) -> f16 transposed (N x Kp), zero pad K (B-side operands)
__global__ void k_cvt_tr(const float* __restrict__ src, _Float16* __restrict__ dst,
                         int K0, int Kp, int Nc)
{
  int i = blockIdx.x * 256 + threadIdx.x;
  if (i >= Nc * Kp) return;
  int n = i / Kp, k = i - n * Kp;
  dst[i] = (k < K0) ? (_Float16)src[(size_t)k * Nc + n] : (_Float16)0.0f;
}

// agent_valid (B,N,T) -> avt (B,T,N)
__global__ void k_avt(const unsigned char* __restrict__ valid, unsigned char* __restrict__ avt)
{
  int i = blockIdx.x * 256 + threadIdx.x;
  if (i >= CB * CT * CN) return;
  int n = i % CN; int bt = i / CN; int t = bt & (CT - 1); int b = bt >> 5;
  avt[i] = valid[((size_t)(b * CN + n)) * CT + t];
}

// agent motion features (B,N,T,4)
__global__ void k_feats(const float* __restrict__ poses, const float* __restrict__ vels,
                        float* __restrict__ out)
{
  int i = blockIdx.x * 256 + threadIdx.x;
  if (i >= CB * CN * CT) return;
  int t = i & (CT - 1);
  const float* p = poses + (size_t)i * 3;
  float px = p[0], py = p[1], hd = p[2];
  float hx = __cosf(hd), hy = __sinf(hd);
  float mx = 0.0f, my = 0.0f;
  if (t > 0) { mx = px - p[-3]; my = py - p[-2]; }
  float vx = vels[(size_t)i * 2], vy = vels[(size_t)i * 2 + 1];
  out[(size_t)i * 4 + 0] = sqrtf(mx * mx + my * my);
  out[(size_t)i * 4 + 1] = atan2f(hx * my - hy * mx, hx * mx + hy * my);
  out[(size_t)i * 4 + 2] = sqrtf(vx * vx + vy * vy);
  out[(size_t)i * 4 + 3] = atan2f(hx * vy - hy * vx, hx * vx + hy * vy);
}

// temporal pair features (B*N, T, T, 4)
__global__ void k_rt_raw(const float* __restrict__ poses, float* __restrict__ out)
{
  int i = blockIdx.x * 256 + threadIdx.x;
  if (i >= CB * CN * CT * CT) return;
  int j  = i & (CT - 1);
  int ii = (i >> 5) & (CT - 1);
  int bn = i >> 10;
  const float* pi = poses + ((size_t)bn * CT + ii) * 3;
  const float* pj = poses + ((size_t)bn * CT + j) * 3;
  float rx = pi[0] - pj[0], ry = pi[1] - pj[1];
  float cx = __cosf(pi[2]) - __cosf(pj[2]);
  float cy = __sinf(pi[2]) - __sinf(pj[2]);
  out[(size_t)i * 4 + 0] = sqrtf(rx * rx + ry * ry);
  out[(size_t)i * 4 + 1] = atan2f(cx * ry - cy * rx, cx * rx + cy * ry);
  out[(size_t)i * 4 + 2] = wrapang(pi[2] - pj[2]);
  out[(size_t)i * 4 + 3] = (float)(ii - j);
}

// KNN agent->polyline: per (b,t,n) select 32 nearest of 128, emit idx + gathered raw rpe feats
__global__ void k_knn_a2pl(const float* __restrict__ poses, const float* __restrict__ plposes,
                           int* __restrict__ knn, float* __restrict__ raw)
{
  int i = blockIdx.x * 256 + threadIdx.x;
  if (i >= CB * CT * CN) return;
  int n = i % CN; int bt = i / CN; int t = bt & (CT - 1); int b = bt >> 5;
  const float* p = poses + (((size_t)(b * CN + n)) * CT + t) * 3;
  float px = p[0], py = p[1], hd = p[2];
  float hx = __cosf(hd), hy = __sinf(hd);
  const float* pl = plposes + (size_t)b * CM * 3;
  unsigned used[4] = {0u, 0u, 0u, 0u};
  for (int kk = 0; kk < CKPL; ++kk) {
    float best = 3.4e38f; int bi = 0;
    for (int m = 0; m < CM; ++m) {
      if (used[m >> 5] & (1u << (m & 31))) continue;
      float dx = px - pl[m * 3], dy = py - pl[m * 3 + 1];
      float dd = dx * dx + dy * dy;
      if (dd < best) { best = dd; bi = m; }
    }
    used[bi >> 5] |= 1u << (bi & 31);
    knn[(size_t)i * CKPL + kk] = bi;
    float dx = px - pl[bi * 3], dy = py - pl[bi * 3 + 1];
    size_t ro = ((size_t)i * CKPL + kk) * 3;
    raw[ro + 0] = sqrtf(dx * dx + dy * dy);
    raw[ro + 1] = atan2f(hx * dy - hy * dx, hx * dx + hy * dy);
    raw[ro + 2] = wrapang(pl[bi * 3 + 2] - hd);
  }
}

// KNN agent->agent: per (b,t,n) select 16 nearest of 48
__global__ void k_knn_a2a(const float* __restrict__ poses,
                          int* __restrict__ knn, float* __restrict__ raw)
{
  int i = blockIdx.x * 256 + threadIdx.x;
  if (i >= CB * CT * CN) return;
  int n = i % CN; int bt = i / CN; int t = bt & (CT - 1); int b = bt >> 5;
  const float* p = poses + (((size_t)(b * CN + n)) * CT + t) * 3;
  float px = p[0], py = p[1], hd = p[2];
  float hx = __cosf(hd), hy = __sinf(hd);
  unsigned long long used = 0ull;
  for (int kk = 0; kk < CKAA; ++kk) {
    float best = 3.4e38f; int bi = 0;
    for (int j = 0; j < CN; ++j) {
      if (used & (1ull << j)) continue;
      const float* q = poses + (((size_t)(b * CN + j)) * CT + t) * 3;
      float dx = px - q[0], dy = py - q[1];
      float dd = dx * dx + dy * dy;
      if (dd < best) { best = dd; bi = j; }
    }
    used |= 1ull << bi;
    knn[(size_t)i * CKAA + kk] = bi;
    const float* q = poses + (((size_t)(b * CN + bi)) * CT + t) * 3;
    float dx = px - q[0], dy = py - q[1];
    size_t ro = ((size_t)i * CKAA + kk) * 3;
    raw[ro + 0] = sqrtf(dx * dx + dy * dy);
    raw[ro + 1] = atan2f(hx * dy - hy * dx, hx * dx + hy * dy);
    raw[ro + 2] = wrapang(hd - q[2]);
  }
}

// fourier front: raw (P,D) dim d -> [cos(2pi f x), sin(2pi f x), x, pad] f16 (P,160)
__global__ void k_four_front(const float* __restrict__ raw, const float* __restrict__ freqs,
                             _Float16* __restrict__ out, int P, int D, int d)
{
  int p = blockIdx.x * 256 + threadIdx.x;
  if (p >= P) return;
  float x = raw[(size_t)p * D + d];
  const float* fr = freqs + d * CF;
  _Float16* o = out + (size_t)p * 160;
#pragma unroll 4
  for (int j = 0; j < CF; j += 2) {
    float f0 = x * fr[j] * TWO_PI_F;
    float f1 = x * fr[j + 1] * TWO_PI_F;
    H2 c, s;
    c.h[0] = (_Float16)__cosf(f0); c.h[1] = (_Float16)__cosf(f1);
    s.h[0] = (_Float16)__sinf(f0); s.h[1] = (_Float16)__sinf(f1);
    *(unsigned*)(o + j)      = c.u;
    *(unsigned*)(o + CF + j) = s.u;
  }
  H2 tail; tail.h[0] = (_Float16)x; tail.h[1] = (_Float16)0.0f;
  *(unsigned*)(o + 128) = tail.u;
#pragma unroll
  for (int j = 130; j < 160; j += 2) *(unsigned*)(o + j) = 0u;
}

// out16 = relu(ln(in + b, g, c)); width 128; one wave per row
__global__ __launch_bounds__(256) void k_bias_ln_relu(
    const float* __restrict__ in, const float* __restrict__ b,
    const float* __restrict__ g, const float* __restrict__ c,
    _Float16* __restrict__ out, int P)
{
  int row = blockIdx.x * 8 + (threadIdx.x >> 5);
  if (row >= P) return;
  int lane = threadIdx.x & 31;
  const float* x = in + (size_t)row * CH;
  float v[4]; float s = 0.0f;
#pragma unroll
  for (int i = 0; i < 4; ++i) { int col = lane + i * 32; v[i] = x[col] + b[col]; s += v[i]; }
  float mean = wred32(s) * (1.0f / 128.0f);
  float q = 0.0f;
#pragma unroll
  for (int i = 0; i < 4; ++i) { float d = v[i] - mean; q += d * d; }
  float rs = rsqrtf(wred32(q) * (1.0f / 128.0f) + 1e-5f);
#pragma unroll
  for (int i = 0; i < 4; ++i) {
    int col = lane + i * 32;
    float y = (v[i] - mean) * rs * g[col] + c[col];
    out[(size_t)row * CH + col] = (_Float16)(y > 0.0f ? y : 0.0f);
  }
}

// fourier tail: y16 = relu(ln(hsum + sum_d b2[d] (+ cat), go, co))
__global__ __launch_bounds__(256) void k_four_post(
    const float* __restrict__ hsum, const float* __restrict__ b2, int D,
    const float* __restrict__ go, const float* __restrict__ co,
    const float* __restrict__ temb, const int* __restrict__ cls, int rpi,
    _Float16* __restrict__ out, int P)
{
  int row = blockIdx.x * 8 + (threadIdx.x >> 5);
  if (row >= P) return;
  int lane = threadIdx.x & 31;
  const float* x = hsum + (size_t)row * CH;
  const float* catp = temb ? (temb + (size_t)cls[row / rpi] * CH) : nullptr;
  float v[4]; float s = 0.0f;
#pragma unroll
  for (int i = 0; i < 4; ++i) {
    int col = lane + i * 32;
    float y = x[col];
    for (int d = 0; d < D; ++d) y += b2[d * CH + col];
    if (catp) y += catp[col];
    v[i] = y; s += y;
  }
  float mean = wred32(s) * (1.0f / 128.0f);
  float q = 0.0f;
#pragma unroll
  for (int i = 0; i < 4; ++i) { float d = v[i] - mean; q += d * d; }
  float rs = rsqrtf(wred32(q) * (1.0f / 128.0f) + 1e-5f);
#pragma unroll
  for (int i = 0; i < 4; ++i) {
    int col = lane + i * 32;
    float y = (v[i] - mean) * rs * go[col] + co[col];
    out[(size_t)row * CH + col] = (_Float16)(y > 0.0f ? y : 0.0f);
  }
}

// residual + LN -> f32 + f16
__global__ __launch_bounds__(256) void k_add_ln(
    const float* __restrict__ xin, const float* __restrict__ dlt,
    const float* __restrict__ g, const float* __restrict__ c,
    float* __restrict__ o32, _Float16* __restrict__ o16, int P)
{
  int row = blockIdx.x * 8 + (threadIdx.x >> 5);
  if (row >= P) return;
  int lane = threadIdx.x & 31;
  float v[4]; float s = 0.0f;
#pragma unroll
  for (int i = 0; i < 4; ++i) {
    int col = lane + i * 32;
    v[i] = xin[(size_t)row * CH + col] + dlt[(size_t)row * CH + col];
    s += v[i];
  }
  float mean = wred32(s) * (1.0f / 128.0f);
  float q = 0.0f;
#pragma unroll
  for (int i = 0; i < 4; ++i) { float d = v[i] - mean; q += d * d; }
  float rs = rsqrtf(wred32(q) * (1.0f / 128.0f) + 1e-5f);
#pragma unroll
  for (int i = 0; i < 4; ++i) {
    int col = lane + i * 32;
    float y = (v[i] - mean) * rs * g[col] + c[col];
    o32[(size_t)row * CH + col] = y;
    o16[(size_t)row * CH + col] = (_Float16)y;
  }
}

// KNN attention core: one block per q-row. logits by 8*K threads, output by 128.
__global__ __launch_bounds__(256) void k_attn_knn(
    const float* __restrict__ q, const float* __restrict__ kmat, const float* __restrict__ vmat,
    const float* __restrict__ rk, const float* __restrict__ rv,
    const int* __restrict__ knn,
    const unsigned char* __restrict__ qvalid, const unsigned char* __restrict__ kvvalid,
    int K, int Nq, int kv_div, int kv_mult,
    _Float16* __restrict__ o)
{
  __shared__ float sl[CNH * 32];
  __shared__ int sidx[32];
  int r = blockIdx.x;
  int bq = r / Nq;
  int kvbase = (bq / kv_div) * kv_mult;
  int t = threadIdx.x;
  bool qv = qvalid[r] != 0;
  if (t < K) sidx[t] = knn[(size_t)r * K + t];
  __syncthreads();
  if (t < CNH * K) {
    int h = t / K, kk = t - h * K;
    int kvrow = kvbase + sidx[kk];
    const float* qp = q + (size_t)r * CH + h * CDH;
    const float* kp = kmat + (size_t)kvrow * CH + h * CDH;
    const float* rp = rk + ((size_t)r * K + kk) * CH + h * CDH;
    float acc = 0.0f;
#pragma unroll
    for (int d = 0; d < CDH; ++d) acc += qp[d] * (kp[d] + rp[d]);
    bool masked = !(qv && (kvvalid[kvrow] != 0));
    sl[h * 32 + kk] = masked ? -1e9f : acc * 0.25f;   // 1/sqrt(16)
  }
  __syncthreads();
  if (t < CH) {
    int h = t >> 4, d = t & 15;
    float mx = -1e30f;
    for (int kk = 0; kk < K; ++kk) mx = fmaxf(mx, sl[h * 32 + kk]);
    float sum = 0.0f, out = 0.0f;
    for (int kk = 0; kk < K; ++kk) {
      float a = __expf(sl[h * 32 + kk] - mx);
      sum += a;
      int kvrow = kvbase + sidx[kk];
      out += a * (vmat[(size_t)kvrow * CH + h * CDH + d] +
                  rv[((size_t)r * K + kk) * CH + h * CDH + d]);
    }
    o[(size_t)r * CH + t] = (_Float16)(out / sum);
  }
}

// dense causal temporal attention core: block per (bq,i), L=T=32
__global__ __launch_bounds__(256) void k_attn_dense(
    const float* __restrict__ q, const float* __restrict__ kmat, const float* __restrict__ vmat,
    const float* __restrict__ rk, const float* __restrict__ rv,
    const unsigned char* __restrict__ valid, _Float16* __restrict__ o)
{
  __shared__ float sl[CNH * 32];
  int r = blockIdx.x;              // = bq*32 + i
  int i = r & 31; int bq = r >> 5;
  int t = threadIdx.x;
  bool qv = valid[r] != 0;
  {
    int h = t >> 5, j = t & 31;
    const float* qp = q + (size_t)r * CH + h * CDH;
    const float* kp = kmat + ((size_t)(bq * CT + j)) * CH + h * CDH;
    const float* rp = rk + ((size_t)r * CT + j) * CH + h * CDH;
    float acc = 0.0f;
#pragma unroll
    for (int d = 0; d < CDH; ++d) acc += qp[d] * (kp[d] + rp[d]);
    bool masked = (!qv) || (j > i);
    sl[h * 32 + j] = masked ? -1e9f : acc * 0.25f;
  }
  __syncthreads();
  if (t < CH) {
    int h = t >> 4, d = t & 15;
    float mx = -1e30f;
    for (int j = 0; j < CT; ++j) mx = fmaxf(mx, sl[h * 32 + j]);
    float sum = 0.0f, out = 0.0f;
    for (int j = 0; j < CT; ++j) {
      float a = __expf(sl[h * 32 + j] - mx);
      sum += a;
      out += a * (vmat[((size_t)(bq * CT + j)) * CH + h * CDH + d] +
                  rv[((size_t)r * CT + j) * CH + h * CDH + d]);
    }
    o[(size_t)r * CH + t] = (_Float16)(out / sum);
  }
}

// (b,i,j,h) -> (b,j,i,h)
__global__ void k_perm(const float* __restrict__ in, float* __restrict__ o32,
                       _Float16* __restrict__ o16, int D0, int D1, int D2, int D3)
{
  size_t idx = (size_t)blockIdx.x * 256 + threadIdx.x;
  size_t tot = (size_t)D0 * D1 * D2 * D3;
  if (idx >= tot) return;
  int h = (int)(idx % D3); size_t rr = idx / D3;
  int j = (int)(rr % D2); rr /= D2;
  int i = (int)(rr % D1); int b = (int)(rr / D1);
  float v = in[idx];
  size_t oo = (((size_t)b * D2 + j) * D1 + i) * (size_t)D3 + h;
  o32[oo] = v;
  if (o16) o16[oo] = (_Float16)v;
}

// ==================== host orchestration ====================
extern "C" void kernel_launch(void* const* d_in, const int* in_sizes, int n_in,
                              void* d_out, int out_size, void* d_ws, size_t ws_size,
                              hipStream_t stream)
{
  (void)in_sizes; (void)out_size; (void)ws_size;
  if (n_in < 136) return;

  struct FP { const float *freqs,*w1,*b1,*g1,*c1,*w2,*b2,*go,*co,*wo,*bo; };
  struct AP { const float *wq,*wk,*wv,*wrk,*wrv,*wo,*g1,*c1,*g2,*c2,*wf1,*bf1,*wf2,*bf2; };
  auto getFP = [&](int s){ FP f;
    f.freqs=(const float*)d_in[s+0]; f.w1=(const float*)d_in[s+1]; f.b1=(const float*)d_in[s+2];
    f.g1=(const float*)d_in[s+3];    f.c1=(const float*)d_in[s+4]; f.w2=(const float*)d_in[s+5];
    f.b2=(const float*)d_in[s+6];    f.go=(const float*)d_in[s+7]; f.co=(const float*)d_in[s+8];
    f.wo=(const float*)d_in[s+9];    f.bo=(const float*)d_in[s+10]; return f; };
  auto getAP = [&](int s){ AP a;
    a.wq=(const float*)d_in[s+0]; a.wk=(const float*)d_in[s+1]; a.wv=(const float*)d_in[s+2];
    a.wrk=(const float*)d_in[s+3]; a.wrv=(const float*)d_in[s+4]; a.wo=(const float*)d_in[s+5];
    a.g1=(const float*)d_in[s+6]; a.c1=(const float*)d_in[s+7]; a.g2=(const float*)d_in[s+8];
    a.c2=(const float*)d_in[s+9]; a.wf1=(const float*)d_in[s+10]; a.bf1=(const float*)d_in[s+11];
    a.wf2=(const float*)d_in[s+12]; a.bf2=(const float*)d_in[s+13]; return a; };

  const float* type_emb = (const float*)d_in[0];
  FP fp_xa = getFP(1), fp_rt = getFP(12), fp_pl = getFP(23), fp_aa = getFP(34);
  AP ap_t[CNL]  = { getAP(45),  getAP(59)  };
  AP ap_pl[CNL] = { getAP(73),  getAP(87)  };
  AP ap_aa[CNL] = { getAP(101), getAP(115) };
  const float* agent_poses = (const float*)d_in[129];
  const float* agent_vels  = (const float*)d_in[130];
  const int*   agent_cls   = (const int*)d_in[131];
  const unsigned char* agent_valid = (const unsigned char*)d_in[132];
  const float* pl_poses    = (const float*)d_in[133];
  const unsigned char* pl_valid = (const unsigned char*)d_in[134];
  const float* x_pl        = (const float*)d_in[135];

  // deterministic bump allocator over d_ws
  char* wp = (char*)d_ws;
  auto alloc = [&](size_t bytes) -> void* {
    void* r = (void*)wp; wp += (bytes + 255) & ~(size_t)255; return r; };

  // B-side weights: store transposed (N x Kp) for contiguous WMMA fragments
  auto cvtT = [&](const float* src, int K0, int Kp, int Nc) -> _Float16* {
    _Float16* dst = (_Float16*)alloc((size_t)Nc * Kp * sizeof(_Float16));
    int tot = Nc * Kp;
    k_cvt_tr<<<(tot + 255) / 256, 256, 0, stream>>>(src, dst, K0, Kp, Nc);
    return dst;
  };
  // A-side activations: plain f16
  auto cvtA = [&](const float* src, int Rows, int Nc) -> _Float16* {
    _Float16* dst = (_Float16*)alloc((size_t)Rows * Nc * sizeof(_Float16));
    int tot = Rows * Nc;
    k_cvt_pad<<<(tot + 255) / 256, 256, 0, stream>>>(src, dst, Rows, Rows, Nc);
    return dst;
  };

  struct FPH { _Float16* w1[4]; _Float16* w2[4]; _Float16* wo; };
  auto makeFPH = [&](const FP& f, int D) {
    FPH h{};
    for (int d = 0; d < D; ++d) {
      h.w1[d] = cvtT(f.w1 + (size_t)d * 129 * CH, 129, 160, CH);  // (128 x 160)
      h.w2[d] = cvtT(f.w2 + (size_t)d * CH * CH, CH, CH, CH);
    }
    h.wo = cvtT(f.wo, CH, CH, CH);
    return h;
  };
  struct APH { _Float16 *wq,*wk,*wv,*wrk,*wrv,*wo,*wf1,*wf2; };
  auto makeAPH = [&](const AP& a) {
    APH h;
    h.wq = cvtT(a.wq,CH,CH,CH);  h.wk = cvtT(a.wk,CH,CH,CH);  h.wv = cvtT(a.wv,CH,CH,CH);
    h.wrk= cvtT(a.wrk,CH,CH,CH); h.wrv= cvtT(a.wrv,CH,CH,CH); h.wo = cvtT(a.wo,CH,CH,CH);
    h.wf1= cvtT(a.wf1,CH,CH,4*CH);      // (512 x 128)
    h.wf2= cvtT(a.wf2,4*CH,4*CH,CH);    // (128 x 512)
    return h;
  };

  FPH fh_xa = makeFPH(fp_xa,4), fh_rt = makeFPH(fp_rt,4), fh_pl = makeFPH(fp_pl,3), fh_aa = makeFPH(fp_aa,3);
  APH ah_t[CNL]  = { makeAPH(ap_t[0]),  makeAPH(ap_t[1])  };
  APH ah_pl[CNL] = { makeAPH(ap_pl[0]), makeAPH(ap_pl[1]) };
  APH ah_aa[CNL] = { makeAPH(ap_aa[0]), makeAPH(ap_aa[1]) };
  _Float16* xpl16 = cvtA(x_pl, CB * CM, CH);

  const int RQ  = CB * CN * CT;          // 3072 (= B*T*N too)
  const int PRT = CB * CN * CT * CT;     // 98304
  const int PPL = CB * CT * CN * CKPL;   // 98304
  const int PAA = CB * CT * CN * CKAA;   // 49152

  float* feats_xa = (float*)alloc((size_t)RQ * 4 * 4);
  float* rt_raw   = (float*)alloc((size_t)PRT * 4 * 4);
  float* pl_raw   = (float*)alloc((size_t)PPL * 3 * 4);
  int*   knn_pl   = (int*)alloc((size_t)RQ * CKPL * 4);
  float* aa_raw   = (float*)alloc((size_t)PAA * 3 * 4);
  int*   knn_aa   = (int*)alloc((size_t)RQ * CKAA * 4);
  unsigned char* avt = (unsigned char*)alloc(RQ);

  _Float16* feat16 = (_Float16*)alloc((size_t)PRT * 160 * 2);
  float*    h32    = (float*)alloc((size_t)PRT * CH * 4);
  _Float16* h16    = (_Float16*)alloc((size_t)PRT * CH * 2);
  float*    hsum   = (float*)alloc((size_t)PRT * CH * 4);

  _Float16* xa16   = (_Float16*)alloc((size_t)RQ * CH * 2);
  _Float16* rt16   = (_Float16*)alloc((size_t)PRT * CH * 2);
  _Float16* rgpl16 = (_Float16*)alloc((size_t)PPL * CH * 2);
  _Float16* rgaa16 = (_Float16*)alloc((size_t)PAA * CH * 2);

  float* q32 = (float*)alloc((size_t)RQ * CH * 4);
  float* k32 = (float*)alloc((size_t)RQ * CH * 4);
  float* v32 = (float*)alloc((size_t)RQ * CH * 4);
  float* rk32 = (float*)alloc((size_t)PRT * CH * 4);
  float* rv32 = (float*)alloc((size_t)PRT * CH * 4);
  _Float16* o16  = (_Float16*)alloc((size_t)RQ * CH * 2);
  float* ows  = (float*)alloc((size_t)RQ * CH * 4);
  float* x1_32 = (float*)alloc((size_t)RQ * CH * 4);
  _Float16* x1_16 = (_Float16*)alloc((size_t)RQ * CH * 2);
  _Float16* ffh16 = (_Float16*)alloc((size_t)RQ * 4 * CH * 2);
  float* ff32 = (float*)alloc((size_t)RQ * CH * 4);
  float* xt32 = (float*)alloc((size_t)RQ * CH * 4);
  _Float16* xt16 = (_Float16*)alloc((size_t)RQ * CH * 2);

  float* x32 = (float*)d_out;   // persistent x_a (B,N,T,H)

  auto gemm = [&](const _Float16* A, const _Float16* Bt, float* Cf, _Float16* Ch,
                  const float* bias, int Mr, int Nc, int Kd, int lda, int ldb,
                  int accum, int relu) {
    k_gemm<<<dim3(Nc / 64, Mr / 64), 128, 0, stream>>>(
        A, Bt, Cf, Ch, bias, Mr, Nc, Kd, lda, ldb, accum, relu);
  };

  auto fourier = [&](const float* raw, int P, int D, const FP& f, const FPH& fh,
                     const float* temb, const int* cls, int rpi,
                     float* out32, _Float16* out16) {
    for (int d = 0; d < D; ++d) {
      k_four_front<<<(P + 255) / 256, 256, 0, stream>>>(raw, f.freqs, feat16, P, D, d);
      gemm(feat16, fh.w1[d], h32, nullptr, nullptr, P, CH, 160, 160, 160, 0, 0);
      k_bias_ln_relu<<<(P + 7) / 8, 256, 0, stream>>>(h32, f.b1 + d * CH, f.g1 + d * CH,
                                                      f.c1 + d * CH, h16, P);
      gemm(h16, fh.w2[d], hsum, nullptr, nullptr, P, CH, CH, CH, CH, d > 0, 0);
    }
    k_four_post<<<(P + 7) / 8, 256, 0, stream>>>(hsum, f.b2, D, f.go, f.co,
                                                 temb, cls, rpi, h16, P);
    gemm(h16, fh.wo, out32, out16, f.bo, P, CH, CH, CH, CH, 0, 0);
  };

  // ---- stage 0: features, masks, KNN, fourier encoders ----
  k_avt<<<(RQ + 255) / 256, 256, 0, stream>>>(agent_valid, avt);
  k_feats<<<(RQ + 255) / 256, 256, 0, stream>>>(agent_poses, agent_vels, feats_xa);
  fourier(feats_xa, RQ, 4, fp_xa, fh_xa, type_emb, agent_cls, CT, x32, xa16);

  k_rt_raw<<<(PRT + 255) / 256, 256, 0, stream>>>(agent_poses, rt_raw);
  fourier(rt_raw, PRT, 4, fp_rt, fh_rt, nullptr, nullptr, 1, nullptr, rt16);

  k_knn_a2pl<<<(RQ + 255) / 256, 256, 0, stream>>>(agent_poses, pl_poses, knn_pl, pl_raw);
  fourier(pl_raw, PPL, 3, fp_pl, fh_pl, nullptr, nullptr, 1, nullptr, rgpl16);

  k_knn_a2a<<<(RQ + 255) / 256, 256, 0, stream>>>(agent_poses, knn_aa, aa_raw);
  fourier(aa_raw, PAA, 3, fp_aa, fh_aa, nullptr, nullptr, 1, nullptr, rgaa16);

  // ---- layers ----
  auto post = [&](const AP& p, const APH& ph, const float* xin32,
                  float* xo32, _Float16* xo16) {
    gemm(o16, ph.wo, ows, nullptr, nullptr, RQ, CH, CH, CH, CH, 0, 0);
    k_add_ln<<<RQ / 8, 256, 0, stream>>>(xin32, ows, p.g1, p.c1, x1_32, x1_16, RQ);
    gemm(x1_16, ph.wf1, nullptr, ffh16, p.bf1, RQ, 4 * CH, CH, CH, CH, 0, 1);
    gemm(ffh16, ph.wf2, ff32, nullptr, p.bf2, RQ, CH, 4 * CH, 4 * CH, 4 * CH, 0, 0);
    k_add_ln<<<RQ / 8, 256, 0, stream>>>(x1_32, ff32, p.g2, p.c2, xo32, xo16, RQ);
  };

  for (int l = 0; l < CNL; ++l) {
    // --- temporal dense attention on (B*N, T, H) == (B,N,T,H) flat ---
    { const AP& p = ap_t[l]; const APH& ph = ah_t[l];
      gemm(xa16, ph.wq, q32, nullptr, nullptr, RQ, CH, CH, CH, CH, 0, 0);
      gemm(xa16, ph.wk, k32, nullptr, nullptr, RQ, CH, CH, CH, CH, 0, 0);
      gemm(xa16, ph.wv, v32, nullptr, nullptr, RQ, CH, CH, CH, CH, 0, 0);
      gemm(rt16, ph.wrk, rk32, nullptr, nullptr, PRT, CH, CH, CH, CH, 0, 0);
      gemm(rt16, ph.wrv, rv32, nullptr, nullptr, PRT, CH, CH, CH, CH, 0, 0);
      k_attn_dense<<<RQ, 256, 0, stream>>>(q32, k32, v32, rk32, rv32, agent_valid, o16);
      post(p, ph, x32, x32, xa16);
    }
    // permute (B,N,T,H) -> (B,T,N,H)
    k_perm<<<((size_t)RQ * CH + 255) / 256, 256, 0, stream>>>(x32, xt32, xt16, CB, CN, CT, CH);

    // --- agent -> polyline KNN attention on (B*T, N, H) ---
    { const AP& p = ap_pl[l]; const APH& ph = ah_pl[l];
      gemm(xt16, ph.wq, q32, nullptr, nullptr, RQ, CH, CH, CH, CH, 0, 0);
      gemm(xpl16, ph.wk, k32, nullptr, nullptr, CB * CM, CH, CH, CH, CH, 0, 0);
      gemm(xpl16, ph.wv, v32, nullptr, nullptr, CB * CM, CH, CH, CH, CH, 0, 0);
      gemm(rgpl16, ph.wrk, rk32, nullptr, nullptr, PPL, CH, CH, CH, CH, 0, 0);
      gemm(rgpl16, ph.wrv, rv32, nullptr, nullptr, PPL, CH, CH, CH, CH, 0, 0);
      k_attn_knn<<<RQ, 256, 0, stream>>>(q32, k32, v32, rk32, rv32, knn_pl,
                                         avt, pl_valid, CKPL, CN, CT, CM, o16);
      post(p, ph, xt32, xt32, xt16);
    }
    // --- agent -> agent KNN attention on (B*T, N, H) ---
    { const AP& p = ap_aa[l]; const APH& ph = ah_aa[l];
      gemm(xt16, ph.wq, q32, nullptr, nullptr, RQ, CH, CH, CH, CH, 0, 0);
      gemm(xt16, ph.wk, k32, nullptr, nullptr, RQ, CH, CH, CH, CH, 0, 0);
      gemm(xt16, ph.wv, v32, nullptr, nullptr, RQ, CH, CH, CH, CH, 0, 0);
      gemm(rgaa16, ph.wrk, rk32, nullptr, nullptr, PAA, CH, CH, CH, CH, 0, 0);
      gemm(rgaa16, ph.wrv, rv32, nullptr, nullptr, PAA, CH, CH, CH, CH, 0, 0);
      k_attn_knn<<<RQ, 256, 0, stream>>>(q32, k32, v32, rk32, rv32, knn_aa,
                                         avt, avt, CKAA, CN, 1, CN, o16);
      post(p, ph, xt32, xt32, xt16);
    }
    // permute back (B,T,N,H) -> (B,N,T,H) into x32 (= d_out) + xa16 for next layer
    k_perm<<<((size_t)RQ * CH + 255) / 256, 256, 0, stream>>>(xt32, x32, xa16, CB, CT, CN, CH);
  }
}
